// DCTLayer_3307124818231
// MI455X (gfx1250) — matrix operand order repaired and checked by reading, not compiled, not run on hardware
//
#include <hip/hip_runtime.h>
#include <math.h>

// ---------------------------------------------------------------------------
// 2-D DCT-II (ortho) via bf16 WMMA with hi/lo split-precision, CDNA5 (gfx1250)
//   out = (log(|C X C^T| + eps) - mean) / std      per (b,c) image, 256x256
// One workgroup (8 wave32) per image; Y = X*C^T staged transposed in LDS as
// pre-split bf16 hi/lo planes (split once at store, zero-convert reads).
// ---------------------------------------------------------------------------

typedef __attribute__((ext_vector_type(16))) __bf16 v16bf;
typedef __attribute__((ext_vector_type(8)))  float  v8f;

#define N256      256
#define LDW       264                        // padded LDS row stride (bf16 elems)
                                             // 528 B/row == 4 (mod 64) dwords ->
                                             // conflict-free b128 access
#define LDS_BYTES (2 * N256 * LDW * 2)       // Yhi + Ylo = 270,336 B <= 320 KB
#define DCT_EPS   1e-13f

#define WMMA_BF16(a, b, c) \
  __builtin_amdgcn_wmma_f32_16x16x32_bf16(false, (a), false, (b), (short)0, (c), false, false)

__device__ __forceinline__ void split_bf16(float x, __bf16 &h, __bf16 &l) {
  h = (__bf16)x;
  l = (__bf16)(x - (float)h);
}

// A fragment (16x32 bf16) from row-major fp32 src (ld = 256), hi/lo split.
// lanes 0-15 : row = row0+lane,    K = k0 + {0..15}
// lanes 16-31: row = row0+lane-16, K = k0 + {16..31}
__device__ __forceinline__ void load_a_frag_f32nt(const float* __restrict__ src,
                                                  int row0, int k0, int lane,
                                                  v16bf &ah, v16bf &al) {
  const float* p = src + (row0 + (lane & 15)) * N256 + k0 + (lane >> 4) * 16;
#pragma unroll
  for (int j = 0; j < 8; ++j) {
    float x0 = __builtin_nontemporal_load(p + j);
    float x1 = __builtin_nontemporal_load(p + 8 + j);
    __bf16 h0, l0, h1, l1;
    split_bf16(x0, h0, l0);
    split_bf16(x1, h1, l1);
    ah[j] = h0;     al[j] = l0;
    ah[8 + j] = h1; al[8 + j] = l1;
  }
}

// A fragment from pre-split bf16 hi/lo row-major tables (ld = 256).
__device__ __forceinline__ void load_a_frag_bf16(const __bf16* __restrict__ hi,
                                                 const __bf16* __restrict__ lo,
                                                 int row0, int k0, int lane,
                                                 v16bf &ah, v16bf &al) {
  const int off = (row0 + (lane & 15)) * N256 + k0 + (lane >> 4) * 16;
  const __bf16* p = hi + off;
  const __bf16* q = lo + off;
#pragma unroll
  for (int j = 0; j < 8; ++j) {
    ah[j] = p[j];     ah[8 + j] = p[8 + j];
    al[j] = q[j];     al[8 + j] = q[8 + j];
  }
}

// B fragment (32x16 bf16) for Y = X*C^T: B[k][n] = C[n][k], so a lane's run is
// contiguous in C's row n.  lanes 0-15: K = k0+0..15; lanes 16-31: K = k0+16..31.
__device__ __forceinline__ void load_b_frag_ct(const __bf16* __restrict__ hi,
                                               const __bf16* __restrict__ lo,
                                               int k0, int n0, int lane,
                                               v16bf &bh, v16bf &bl) {
  const int off = (n0 + (lane & 15)) * N256 + k0 + (lane >> 4) * 16;
  const __bf16* p = hi + off;
  const __bf16* q = lo + off;
#pragma unroll
  for (int e = 0; e < 16; ++e) { bh[e] = p[e]; bl[e] = q[e]; }
}

// B fragment for out = C*Y from LDS: B[k=i][n=w] = Y[i][w] = Yt[w][i].
// Pre-split planes -> pure ds_load_b128 traffic, no conversion VALU.
__device__ __forceinline__ void load_b_frag_lds(const __bf16* __restrict__ Yhi,
                                                const __bf16* __restrict__ Ylo,
                                                int k0, int n0, int lane,
                                                v16bf &bh, v16bf &bl) {
  const int off = (n0 + (lane & 15)) * LDW + k0 + (lane >> 4) * 16;
  const __bf16* p = Yhi + off;
  const __bf16* q = Ylo + off;
#pragma unroll
  for (int e = 0; e < 16; ++e) { bh[e] = p[e]; bl[e] = q[e]; }
}

// Store 16x16 fp32 accumulator tile transposed into LDS as bf16 hi/lo planes.
// D layout: element e: lanes 0-15 -> M = e, N = lane; lanes 16-31 -> M = 8+e.
__device__ __forceinline__ void store_y_lds(__bf16* __restrict__ Yhi,
                                            __bf16* __restrict__ Ylo,
                                            const v8f &acc,
                                            int m0, int n0, int lane) {
  const int off = (n0 + (lane & 15)) * LDW + m0 + (lane >> 4) * 8;
  __bf16* ph = Yhi + off;
  __bf16* pl = Ylo + off;
#pragma unroll
  for (int e = 0; e < 8; ++e) {
    __bf16 h, l;
    split_bf16(acc[e], h, l);
    ph[e] = h;
    pl[e] = l;
  }
}

// ---------------------------------------------------------------------------
// Setup: orthonormal DCT-II basis, pre-split into bf16 hi/lo tables.
// C[k][i] = s_k * cos(pi*(2i+1)*k / 512), s_0 = 1/16, s_k = sqrt(2)/16.
// ---------------------------------------------------------------------------
__global__ void dct_basis_kernel(__bf16* __restrict__ Chi, __bf16* __restrict__ Clo) {
  const int k = blockIdx.x;
  const int i = threadIdx.x;
  const double PI = 3.14159265358979323846;
  const float scale = (k == 0) ? 0.0625f : 0.08838834764831845f;
  const double ang = PI * (double)((2 * i + 1) * k) / 512.0;
  const float c = scale * (float)cos(ang);
  __bf16 h, l;
  split_bf16(c, h, l);
  Chi[k * N256 + i] = h;
  Clo[k * N256 + i] = l;
}

// ---------------------------------------------------------------------------
// Main: one image per workgroup.
// ---------------------------------------------------------------------------
__global__ void __launch_bounds__(256)
dct2_wmma_kernel(const float* __restrict__ X,
                 const float* __restrict__ mean_p,
                 const float* __restrict__ std_p,
                 const __bf16* __restrict__ Chi,
                 const __bf16* __restrict__ Clo,
                 float* __restrict__ out) {
  extern __shared__ __bf16 Ybuf[];              // Yhi[256][LDW] ++ Ylo[256][LDW]
  __bf16* Yhi = Ybuf;
  __bf16* Ylo = Ybuf + N256 * LDW;

  const int img  = blockIdx.x;
  const float* xi = X   + (size_t)img * (N256 * N256);
  float*       oi = out + (size_t)img * (N256 * N256);
  const int lane = threadIdx.x & 31;
  const int wave = threadIdx.x >> 5;

  const float mean    = *mean_p;
  const float inv_std = 1.0f / (*std_p);

  // ---------------- Stage A: Y = X * C^T, stored transposed in LDS ---------
#pragma unroll
  for (int t = 0; t < 2; ++t) {
    const int m0 = (wave * 2 + t) * 16;           // 16 rows of X per tile
    v16bf axh[8], axl[8];                         // X read from HBM exactly once
#pragma unroll
    for (int ks = 0; ks < 8; ++ks)
      load_a_frag_f32nt(xi, m0, ks * 32, lane, axh[ks], axl[ks]);

    for (int nt = 0; nt < 16; ++nt) {
      v8f acc = {0.f, 0.f, 0.f, 0.f, 0.f, 0.f, 0.f, 0.f};
#pragma unroll
      for (int ks = 0; ks < 8; ++ks) {
        v16bf bh, bl;
        load_b_frag_ct(Chi, Clo, ks * 32, nt * 16, lane, bh, bl);
        acc = WMMA_BF16(axh[ks], bh, acc);        // hi*hi
        acc = WMMA_BF16(axh[ks], bl, acc);        // hi*lo
        acc = WMMA_BF16(axl[ks], bh, acc);        // lo*hi
      }
      store_y_lds(Yhi, Ylo, acc, m0, nt * 16, lane);
    }
  }

  __syncthreads();

  // ---------------- Stage B: out = C * Y, fused log-normalize epilogue -----
#pragma unroll
  for (int t = 0; t < 2; ++t) {
    const int h0 = (wave * 2 + t) * 16;           // 16 output rows per tile
    v16bf ach[8], acl[8];                         // basis rows (L2-resident)
#pragma unroll
    for (int ks = 0; ks < 8; ++ks)
      load_a_frag_bf16(Chi, Clo, h0, ks * 32, lane, ach[ks], acl[ks]);

    for (int nt = 0; nt < 16; ++nt) {
      v8f acc = {0.f, 0.f, 0.f, 0.f, 0.f, 0.f, 0.f, 0.f};
#pragma unroll
      for (int ks = 0; ks < 8; ++ks) {
        v16bf bh, bl;
        load_b_frag_lds(Yhi, Ylo, ks * 32, nt * 16, lane, bh, bl);
        acc = WMMA_BF16(ach[ks], bh, acc);
        acc = WMMA_BF16(ach[ks], bl, acc);
        acc = WMMA_BF16(acl[ks], bh, acc);
      }
      // epilogue + streaming store (don't pollute L2 with write-once output)
      const int w  = nt * 16 + (lane & 15);
      const int hb = h0 + (lane >> 4) * 8;
#pragma unroll
      for (int e = 0; e < 8; ++e) {
        float v = (logf(fabsf(acc[e]) + DCT_EPS) - mean) * inv_std;
        __builtin_nontemporal_store(v, oi + (size_t)(hb + e) * N256 + w);
      }
    }
  }
}

// ---------------------------------------------------------------------------
extern "C" void kernel_launch(void* const* d_in, const int* in_sizes, int n_in,
                              void* d_out, int out_size, void* d_ws, size_t ws_size,
                              hipStream_t stream) {
  const float* X      = (const float*)d_in[0];   // [128,3,256,256] fp32
  const float* mean_p = (const float*)d_in[1];   // scalar
  const float* std_p  = (const float*)d_in[2];   // scalar
  float* out = (float*)d_out;

  // workspace: bf16 hi/lo split of the 256x256 DCT basis (128 KB + 128 KB)
  __bf16* Chi = (__bf16*)d_ws;
  __bf16* Clo = Chi + N256 * N256;

  dct_basis_kernel<<<N256, N256, 0, stream>>>(Chi, Clo);

  (void)hipFuncSetAttribute((const void*)dct2_wmma_kernel,
                            hipFuncAttributeMaxDynamicSharedMemorySize, LDS_BYTES);
  dct2_wmma_kernel<<<128 * 3, 256, LDS_BYTES, stream>>>(X, mean_p, std_p, Chi, Clo, out);
}
